// DeepAHS_61735859912786
// MI455X (gfx1250) — matrix operand order repaired and checked by reading, not compiled, not run on hardware
//
#include <hip/hip_runtime.h>
#include <math.h>

typedef __attribute__((ext_vector_type(16))) _Float16 v16h;
typedef __attribute__((ext_vector_type(8)))  _Float16 v8h;
typedef __attribute__((ext_vector_type(8)))  float    v8f;

// Problem constants
#define BN      4
#define LLEN    96064
#define NFFT    128
#define HOP     64
#define TFRAMES 1500
#define FBINS   65
#define HID     257
#define G3      771          // 3*HID
#define EPSV    1e-8f
#define PI_F    3.14159265358979323846f

// GEMM padding
#define KPAD    288          // 9 slabs of 32 (K = 257)
#define KPAD2   96           // 3 slabs of 32 (K = 65)
#define NTILES  49           // ceil(771/16)
#define NROWPAD 784          // 49*16

// ---- workspace layout (bytes) ----
static constexpr size_t alignUp(size_t x) { return (x + 4095) & ~size_t(4095); }
static constexpr size_t SZ_XK   = (size_t)BN * 2 * TFRAMES * FBINS * 4;    // 3.12 MB
static constexpr size_t SZ_X16  = (size_t)BN * TFRAMES * KPAD2 * 2;        // 1.15 MB
static constexpr size_t SZ_XG   = (size_t)BN * TFRAMES * G3 * 4;           // 18.5 MB
static constexpr size_t SZ_WHH  = (size_t)NROWPAD * KPAD * 2;              // 451 KB
static constexpr size_t SZ_WIH  = (size_t)NROWPAD * KPAD2 * 2;             // 150 KB
static constexpr size_t OFF_XK  = 0;
static constexpr size_t OFF_XKR = OFF_XK  + alignUp(SZ_XK);
static constexpr size_t OFF_X16 = OFF_XKR + alignUp(SZ_XK);
static constexpr size_t OFF_XG  = OFF_X16 + alignUp(SZ_X16);
static constexpr size_t OFF_WHH = OFF_XG  + alignUp(SZ_XG);
static constexpr size_t OFF_WIH = OFF_WHH + alignUp(SZ_WHH);

// ------------------------------------------------------------------
// 1) STFT as windowed 128-pt DFT with LDS twiddle tables.
// ------------------------------------------------------------------
__global__ void stft_kernel(const float* __restrict__ inp,
                            const float* __restrict__ ref,
                            float* __restrict__ Xk, float* __restrict__ Xkr) {
  int fid   = blockIdx.x;                 // 0 .. 2*BN*TFRAMES-1
  int which = fid / (BN * TFRAMES);
  int bt    = fid % (BN * TFRAMES);
  int b = bt / TFRAMES, t = bt % TFRAMES;
  const float* src = which ? ref : inp;
  float*       dst = which ? Xkr : Xk;

  __shared__ float fr[NFFT], twc[NFFT], tws[NFFT];
  int tid = threadIdx.x;
  if (tid < NFFT) {
    float n = (float)tid;
    float win = sinf(PI_F * n / (float)NFFT);          // sqrt-hann
    fr[tid] = src[(size_t)b * LLEN + (size_t)t * HOP + tid] * win;
    float ang = 2.0f * PI_F * n / (float)NFFT;
    twc[tid] = cosf(ang);
    tws[tid] = sinf(ang);
  }
  __syncthreads();
  if (tid < FBINS) {
    float re = 0.f, im = 0.f;
    for (int n = 0; n < NFFT; ++n) {
      int k = (tid * n) & (NFFT - 1);
      re += fr[n] * twc[k];
      im -= fr[n] * tws[k];
    }
    size_t base = (((size_t)b * 2 + 0) * TFRAMES + t) * FBINS + tid;
    dst[base] = re;
    dst[base + (size_t)TFRAMES * FBINS] = im;
  }
}

// ------------------------------------------------------------------
// 2) Fused features + fc1 -> padded f16 x16 (B*T, 96).
//    cov collapses to a real scalar; duplicated corr slices get
//    pre-summed fc1 weights; 368-dim features never materialized.
// ------------------------------------------------------------------
__global__ void feat_fc1_kernel(const float* __restrict__ Xk,
                                const float* __restrict__ Xkr,
                                const float* __restrict__ fc1_w,
                                const float* __restrict__ fc1_b,
                                _Float16* __restrict__ x16) {
  __shared__ float w[368];
  int tid = threadIdx.x;
  for (int i = tid; i < 368; i += blockDim.x) w[i] = fc1_w[i];
  __syncthreads();

  int idx = blockIdx.x * blockDim.x + tid;
  if (idx >= BN * TFRAMES * KPAD2) return;
  int f  = idx % KPAD2;
  if (f >= FBINS) { x16[idx] = (_Float16)0.f; return; }   // K padding
  int bt = idx / KPAD2;
  int t  = bt % TFRAMES;
  int b  = bt / TFRAMES;

  auto at = [&](const float* X, int c, int tt, int ff) -> float {
    return X[(((size_t)b * 2 + c) * TFRAMES + tt) * FBINS + ff];
  };

  float re  = at(Xk, 0, t, f),  im  = at(Xk, 1, t, f);
  float rer = at(Xkr, 0, t, f), imr = at(Xkr, 1, t, f);

  float acc = fc1_b[0];
  acc += w[0] * logf(re * re + im * im + EPSV);
  acc += w[1] * logf(rer * rer + imr * imr + EPSV);
  float a = 0.5f * (re - rer);                 // collapsed channel covariance
  acc += (w[2] - w[3] + w[4]) * (a * a);

  for (int c = 0; c < 2; ++c) {                // time-corr of mic STFT
    float v[9]; float e = 0.f;
    for (int i = 0; i < 9; ++i) {
      int ti = t - 8 + i;
      float vv = (ti >= 0) ? at(Xk, c, ti, f) : 0.0f;
      v[i] = vv; e += vv * vv;
    }
    float inv = 1.0f / (e + EPSV);
    const float* w1 = w + 8  + c * 45;
    const float* w2 = w + 98 + c * 45;
    int d = 0;
    for (int ii = 0; ii < 9; ++ii)
      for (int jj = 0; jj <= ii; ++jj) {
        acc += (w1[d] + w2[d]) * (v[ii] * v[jj] * inv); ++d;
      }
  }
  for (int c = 0; c < 2; ++c) {                // freq-corr of ref STFT
    float v[9]; float e = 0.f;
    for (int i = 0; i < 9; ++i) {
      int fi = f - 8 + i;
      float vv = (fi >= 0) ? at(Xkr, c, t, fi) : 0.0f;
      v[i] = vv; e += vv * vv;
    }
    float inv = 1.0f / (e + EPSV);
    const float* w1 = w + 188 + c * 45;
    const float* w2 = w + 278 + c * 45;
    int d = 0;
    for (int ii = 0; ii < 9; ++ii)
      for (int jj = 0; jj <= ii; ++jj) {
        acc += (w1[d] + w2[d]) * (v[ii] * v[jj] * inv); ++d;
      }
  }
  x16[idx] = (_Float16)acc;
}

// ------------------------------------------------------------------
// 3) Zero-padded f16 weight conversions
// ------------------------------------------------------------------
__global__ void whh_cvt_kernel(const float* __restrict__ w_hh,
                               _Float16* __restrict__ whh16) {
  int idx = blockIdx.x * blockDim.x + threadIdx.x;
  if (idx >= NROWPAD * KPAD) return;
  int rr = idx / KPAD, cc = idx % KPAD;
  float v = (rr < G3 && cc < HID) ? w_hh[rr * HID + cc] : 0.0f;
  whh16[idx] = (_Float16)v;
}

__global__ void wih_cvt_kernel(const float* __restrict__ w_ih,
                               _Float16* __restrict__ wih16) {
  int idx = blockIdx.x * blockDim.x + threadIdx.x;
  if (idx >= NROWPAD * KPAD2) return;
  int rr = idx / KPAD2, cc = idx % KPAD2;
  float v = (rr < G3 && cc < FBINS) ? w_ih[rr * FBINS + cc] : 0.0f;
  wih16[idx] = (_Float16)v;
}

// ---- fragment helpers (ISA 7.12.2 layouts), branch-free 16B loads ----
__device__ __forceinline__ v16h load_a_frag(const _Float16* row, int slab, int lane) {
  int base = slab * 32 + ((lane >= 16) ? 8 : 0);
  v8h lo = *(const v8h*)(row + base);        // K = base .. base+7
  v8h hi = *(const v8h*)(row + base + 16);   // K = base+16 .. base+23
  return __builtin_shufflevector(lo, hi, 0, 1, 2, 3, 4, 5, 6, 7,
                                 8, 9, 10, 11, 12, 13, 14, 15);
}
__device__ __forceinline__ v16h load_b_frag(const _Float16* row, int slab, int khalf) {
  int base = slab * 32 + khalf * 8;
  v8h lo = *(const v8h*)(row + base);        // K = base .. base+7
  v8h hi = *(const v8h*)(row + base + 16);   // K = base+16 .. base+23
  return __builtin_shufflevector(lo, hi, 0, 1, 2, 3, 4, 5, 6, 7,
                                 8, 9, 10, 11, 12, 13, 14, 15);
}

// ------------------------------------------------------------------
// 4) xg = x @ w_ih^T + b_ih via v_wmma_f32_16x16x32_f16
// ------------------------------------------------------------------
__global__ void xg_gemm_kernel(const _Float16* __restrict__ x16,
                               const _Float16* __restrict__ wih16,
                               const float* __restrict__ b_ih,
                               float* __restrict__ xg) {
  int wid  = (blockIdx.x * blockDim.x + threadIdx.x) >> 5;
  int lane = threadIdx.x & 31;
  const int MT = (BN * TFRAMES) / 16;   // 375, exact
  if (wid >= MT * NTILES) return;       // wave-uniform
  int mt = wid / NTILES, nt = wid % NTILES;
  const _Float16* arow = x16   + (size_t)(mt * 16 + (lane & 15)) * KPAD2;
  const _Float16* brow = wih16 + (size_t)(nt * 16 + (lane & 15)) * KPAD2;
  int khalf = lane >> 4;

  v8f acc = {};
#pragma unroll
  for (int slab = 0; slab < 3; ++slab) {
    v16h af = load_a_frag(arow, slab, lane);
    v16h bf = load_b_frag(brow, slab, khalf);
    acc = __builtin_amdgcn_wmma_f32_16x16x32_f16(false, af, false, bf,
                                                 (short)0, acc, false, false);
  }
  int n = nt * 16 + (lane & 15);
#pragma unroll
  for (int v = 0; v < 8; ++v) {
    int mg = mt * 16 + v + 8 * khalf;
    if (n < G3) xg[(size_t)mg * G3 + n] = acc[v] + b_ih[n];
  }
}

// ------------------------------------------------------------------
// 5) Persistent single-WGP GRU scan. gh = h @ w_hh^T via 49x9 WMMAs
//    over 32 waves; gate inputs (xg rows) staged one step ahead into
//    double-buffered LDS with GLOBAL_LOAD_ASYNC_TO_LDS_B32 (ASYNCcnt).
// ------------------------------------------------------------------
#define XGTOT   (BN * G3)        // 3084
#define XGASYNC 3072             // 3 async b32 per thread (1024 thr)

__global__ void __launch_bounds__(1024, 1)
gru_kernel(const float* __restrict__ xg, const _Float16* __restrict__ whh16,
           const float* __restrict__ b_hh, float* __restrict__ out) {
  __shared__ float hS[BN * HID];                                  // hidden state
  __shared__ __attribute__((aligned(16))) _Float16 hA[16 * KPAD]; // f16 A matrix
  __shared__ float ghS[BN * NROWPAD];                             // gh + b_hh
  __shared__ float bhhS[NROWPAD];
  __shared__ float xgS[2][XGTOT];                                 // staged gate inputs

  int tid  = threadIdx.x;
  int nthr = blockDim.x;
  for (int i = tid; i < BN * HID; i += nthr) hS[i] = 0.f;
  for (int i = tid; i < 16 * KPAD; i += nthr) hA[i] = (_Float16)0.f;
  for (int i = tid; i < NROWPAD; i += nthr) bhhS[i] = (i < G3) ? b_hh[i] : 0.f;
  __syncthreads();

  int wave = tid >> 5, lane = tid & 31;
  int nwaves = nthr >> 5;
  const int arow  = lane & 15;
  const int khalf = lane >> 4;

  // issue exactly 3 async b32 loads per thread for step `tn` into buf
  auto issue_async = [&](int tn) {
    int buf = tn & 1;
#pragma unroll
    for (int k = 0; k < 3; ++k) {
      int e = tid + k * 1024;                 // e < 3072
      int b = e / G3, col = e % G3;
      const float* g = xg + ((size_t)b * TFRAMES + tn) * G3 + col;
      unsigned int l = (unsigned int)(size_t)&xgS[buf][e];
      asm volatile("global_load_async_to_lds_b32 %0, %1, off"
                   :: "v"(l), "v"(g) : "memory");
    }
  };

  issue_async(0);   // prologue: stage t = 0

  for (int t = 0; t < TFRAMES; ++t) {
    if (t + 1 < TFRAMES) issue_async(t + 1);   // overlap with this step's GEMM

    // (a) refresh f16 copy of h
    for (int i = tid; i < BN * HID; i += nthr) {
      int b = i / HID, k = i % HID;
      hA[b * KPAD + k] = (_Float16)hS[i];
    }
    __syncthreads();

    // (b) A-fragments for all 9 K-slabs, kept in VGPRs across tiles
    const _Float16* hrow = hA + arow * KPAD;
    v16h af[9];
#pragma unroll
    for (int slab = 0; slab < 9; ++slab) af[slab] = load_a_frag(hrow, slab, lane);

    // (c) gh N-tiles
    for (int tile = wave; tile < NTILES; tile += nwaves) {
      int bcol = tile * 16 + (lane & 15);
      const _Float16* brow = whh16 + (size_t)bcol * KPAD;
      v8f acc = {};
#pragma unroll
      for (int slab = 0; slab < 9; ++slab) {
        v16h bf = load_b_frag(brow, slab, khalf);
        acc = __builtin_amdgcn_wmma_f32_16x16x32_f16(false, af[slab], false, bf,
                                                     (short)0, acc, false, false);
      }
      if (lane < 16) {                         // rows 0..3 are the real batches
#pragma unroll
        for (int v = 0; v < 4; ++v)
          if (bcol < G3) ghS[v * NROWPAD + bcol] = acc[v] + bhhS[bcol];
      }
    }

    // drain this step's staged xg (3 newer loads may stay in flight)
    if (t + 1 < TFRAMES) asm volatile("s_wait_asynccnt 0x3" ::: "memory");
    else                 asm volatile("s_wait_asynccnt 0x0" ::: "memory");
    __syncthreads();

    // (d) fused gate math + output store
    {
      const float* xs = xgS[t & 1];
      size_t xgbase = (size_t)t * G3;          // + b*TFRAMES*G3 per batch
      for (int i = tid; i < BN * HID; i += nthr) {
        int b = i / HID, j = i % HID;
        int er = b * G3 + j;
        int ez = er + HID;
        int en = ez + HID;
        size_t xgrow = ((size_t)b * TFRAMES) * G3 + xgbase;
        float r  = (er < XGASYNC ? xs[er] : xg[xgrow + j])           + ghS[b * NROWPAD + j];
        float z  = (ez < XGASYNC ? xs[ez] : xg[xgrow + HID + j])     + ghS[b * NROWPAD + HID + j];
        float xn = (en < XGASYNC ? xs[en] : xg[xgrow + 2 * HID + j]);
        float hn = ghS[b * NROWPAD + 2 * HID + j];
        r = 1.f / (1.f + expf(-r));
        z = 1.f / (1.f + expf(-z));
        float nn = tanhf(xn + r * hn);
        float h  = (1.f - z) * nn + z * hS[i];
        hS[i] = h;
        out[((size_t)b * TFRAMES + t) * HID + j] = h;
      }
    }
    __syncthreads();
  }
}

// ------------------------------------------------------------------
extern "C" void kernel_launch(void* const* d_in, const int* in_sizes, int n_in,
                              void* d_out, int out_size, void* d_ws, size_t ws_size,
                              hipStream_t stream) {
  const float* inp   = (const float*)d_in[0];
  const float* ref   = (const float*)d_in[1];
  const float* fc1_w = (const float*)d_in[2];
  const float* fc1_b = (const float*)d_in[3];
  const float* w_ih  = (const float*)d_in[4];
  const float* w_hh  = (const float*)d_in[5];
  const float* b_ih  = (const float*)d_in[6];
  const float* b_hh  = (const float*)d_in[7];
  float* out = (float*)d_out;

  char* ws = (char*)d_ws;
  float*    Xk    = (float*)(ws + OFF_XK);
  float*    Xkr   = (float*)(ws + OFF_XKR);
  _Float16* x16   = (_Float16*)(ws + OFF_X16);
  float*    xg    = (float*)(ws + OFF_XG);
  _Float16* whh16 = (_Float16*)(ws + OFF_WHH);
  _Float16* wih16 = (_Float16*)(ws + OFF_WIH);

  // 1) STFT of both signals
  stft_kernel<<<2 * BN * TFRAMES, 128, 0, stream>>>(inp, ref, Xk, Xkr);

  // 2) fused features + fc1 -> padded f16 x16
  {
    int total = BN * TFRAMES * KPAD2;
    feat_fc1_kernel<<<(total + 255) / 256, 256, 0, stream>>>(Xk, Xkr, fc1_w, fc1_b, x16);
  }

  // 3) padded f16 weights
  {
    int total = NROWPAD * KPAD;
    whh_cvt_kernel<<<(total + 255) / 256, 256, 0, stream>>>(w_hh, whh16);
    total = NROWPAD * KPAD2;
    wih_cvt_kernel<<<(total + 255) / 256, 256, 0, stream>>>(w_ih, wih16);
  }

  // 4) xg = x @ w_ih^T + b_ih (WMMA GEMM)
  {
    int waves  = ((BN * TFRAMES) / 16) * NTILES;   // 18375
    int blocks = (waves * 32 + 255) / 256;
    xg_gemm_kernel<<<blocks, 256, 0, stream>>>(x16, wih16, b_ih, xg);
  }

  // 5) persistent GRU scan (single WGP, 32 waves, async-staged gates)
  gru_kernel<<<1, 1024, 0, stream>>>(xg, whh16, b_hh, out);
}
